// DIFFPOOL_AE_32392643346839
// MI455X (gfx1250) — compile-verified, matrix operand order checked by reading, not run on hardware
//
#include <hip/hip_runtime.h>
#include <hip/hip_bf16.h>
#include <stdint.h>

// ===========================================================================
// DiffPool graph autoencoder for MI455X (gfx1250, wave32, WMMA).
// Dense GEMMs: v_wmma_f32_16x16x32_f16, 64x64 C-tile / 128 threads (4 waves),
// BLOCK_K=64, double-buffered LDS (1 barrier per 8 WMMAs), pipelined
// float4 global fills (all 8 loads in flight before conversion).
// Sparse GCN aggregation: f32 global atomics.
// ===========================================================================

#define N_NODES 4096
#define FEATD   128
#define HIDD    256
#define KCL     1024

typedef __attribute__((ext_vector_type(16))) _Float16 v16h;
typedef __attribute__((ext_vector_type(8)))  float    v8f;

#define LDP 72   // LDS row stride in halfwords: 64 + 8 pad (144B, conflict-free)

// WMMA 16-bit A/B fragment load from a 32-halfword LDS window (row-major,
// one matrix row per LDS row). ISA 7.12.2: lane -> m = lane%16, half = lane/16;
// element i -> K = (i/8)*16 + 8*half + (i%8); pairs are K-consecutive dwords.
__device__ __forceinline__ v16h ld_frag(const _Float16* rowp, int half) {
  const uint32_t* p = (const uint32_t*)rowp;
  union { v16h v; uint32_t u[8]; } f;
  const int o = half << 2;
  f.u[0] = p[o + 0]; f.u[1] = p[o + 1]; f.u[2] = p[o + 2]; f.u[3] = p[o + 3];
  f.u[4] = p[o + 8]; f.u[5] = p[o + 9]; f.u[6] = p[o + 10]; f.u[7] = p[o + 11];
  return f.v;
}

// ---------------------------------------------------------------------------
// Tiled WMMA GEMM: C[M,Nc] = op(A) @ op(B)
//   TA=1: A stored [Kd,M] (use A^T).  TB=1: B stored [Nc,Kd] (use B^T).
//   EPI=0: C = acc
//   EPI=1: C = rs[m]*acc + (bias ? bias[n] : 0)
//   EPI=2: link loss: accumulate (adj - acc)^2 (diag from adjDiag) -> linkAcc
// Requires M%64==0, Nc%64==0, Kd%64==0 (true for every call here).
// ---------------------------------------------------------------------------
template<int TA, int TB, int EPI>
__global__ __launch_bounds__(128)
void gemm_wmma_kernel(const float* __restrict__ Aptr, const float* __restrict__ Bptr,
                      float* __restrict__ Cptr, int M, int Nc, int Kd,
                      const float* __restrict__ rs, const float* __restrict__ bias,
                      const float* __restrict__ adjA, const float* __restrict__ adjDiag,
                      float* __restrict__ linkAcc, int adjN) {
  __shared__ __align__(16) _Float16 As[2][64 * LDP];
  __shared__ __align__(16) _Float16 Bs[2][64 * LDP];
  __shared__ float red[128];

  const int tid  = threadIdx.x;
  const int lane = tid & 31;
  const int wave = tid >> 5;
  const int wr   = (wave >> 1) << 5;   // wave row offset in 64-tile
  const int wc   = (wave & 1) << 5;    // wave col offset in 64-tile
  const int m0   = blockIdx.y << 6;
  const int n0   = blockIdx.x << 6;

  const int lda = TA ? M  : Kd;
  const int ldb = TB ? Kd : Nc;

  // Stage one 64x64 f32->f16 tile pair into LDS buffer `bufi` at k-offset kk.
  // Phase 1 issues all 8 float4 loads per matrix (kept in flight), phase 2
  // converts + stores; the contiguous axis of each layout gets the vector load.
  auto fill_tile = [&](int bufi, int kk) {
    _Float16* Asb = &As[bufi][0];
    _Float16* Bsb = &Bs[bufi][0];
    float4 va[8];
    // ---- A ----
    if (!TA) {  // A row-major [M,Kd]: contiguous along k
#pragma unroll
      for (int j = 0; j < 8; ++j) {
        const int g = tid + (j << 7);
        const int cq = (g & 15) << 2, r = g >> 4;
        va[j] = *(const float4*)&Aptr[(size_t)(m0 + r) * lda + kk + cq];
      }
#pragma unroll
      for (int j = 0; j < 8; ++j) {
        const int g = tid + (j << 7);
        const int cq = (g & 15) << 2, r = g >> 4;
        union { _Float16 h[4]; uint2 u; } pk;
        pk.h[0] = (_Float16)va[j].x; pk.h[1] = (_Float16)va[j].y;
        pk.h[2] = (_Float16)va[j].z; pk.h[3] = (_Float16)va[j].w;
        *(uint2*)&Asb[r * LDP + cq] = pk.u;
      }
    } else {    // A stored [Kd,M]: contiguous along m -> scattered LDS stores
#pragma unroll
      for (int j = 0; j < 8; ++j) {
        const int g = tid + (j << 7);
        const int rq = (g & 15) << 2, c = g >> 4;
        va[j] = *(const float4*)&Aptr[(size_t)(kk + c) * lda + m0 + rq];
      }
#pragma unroll
      for (int j = 0; j < 8; ++j) {
        const int g = tid + (j << 7);
        const int rq = (g & 15) << 2, c = g >> 4;
        Asb[(rq + 0) * LDP + c] = (_Float16)va[j].x;
        Asb[(rq + 1) * LDP + c] = (_Float16)va[j].y;
        Asb[(rq + 2) * LDP + c] = (_Float16)va[j].z;
        Asb[(rq + 3) * LDP + c] = (_Float16)va[j].w;
      }
    }
    // ---- B ----
    if (TB) {   // B stored [Nc,Kd]: Bs[n][k] contiguous along k
#pragma unroll
      for (int j = 0; j < 8; ++j) {
        const int g = tid + (j << 7);
        const int cq = (g & 15) << 2, r = g >> 4;
        va[j] = *(const float4*)&Bptr[(size_t)(n0 + r) * ldb + kk + cq];
      }
#pragma unroll
      for (int j = 0; j < 8; ++j) {
        const int g = tid + (j << 7);
        const int cq = (g & 15) << 2, r = g >> 4;
        union { _Float16 h[4]; uint2 u; } pk;
        pk.h[0] = (_Float16)va[j].x; pk.h[1] = (_Float16)va[j].y;
        pk.h[2] = (_Float16)va[j].z; pk.h[3] = (_Float16)va[j].w;
        *(uint2*)&Bsb[r * LDP + cq] = pk.u;
      }
    } else {    // B stored [Kd,Nc]: contiguous along n -> scattered LDS stores
#pragma unroll
      for (int j = 0; j < 8; ++j) {
        const int g = tid + (j << 7);
        const int rq = (g & 15) << 2, c = g >> 4;
        va[j] = *(const float4*)&Bptr[(size_t)(kk + c) * ldb + n0 + rq];
      }
#pragma unroll
      for (int j = 0; j < 8; ++j) {
        const int g = tid + (j << 7);
        const int rq = (g & 15) << 2, c = g >> 4;
        Bsb[(rq + 0) * LDP + c] = (_Float16)va[j].x;
        Bsb[(rq + 1) * LDP + c] = (_Float16)va[j].y;
        Bsb[(rq + 2) * LDP + c] = (_Float16)va[j].z;
        Bsb[(rq + 3) * LDP + c] = (_Float16)va[j].w;
      }
    }
  };

  const v8f vzero = {0.f, 0.f, 0.f, 0.f, 0.f, 0.f, 0.f, 0.f};
  v8f acc[2][2];
  acc[0][0] = vzero; acc[0][1] = vzero; acc[1][0] = vzero; acc[1][1] = vzero;

  const int half = lane >> 4;
  const int mrow = lane & 15;
  const int nk   = Kd >> 6;

  fill_tile(0, 0);
  int buf = 0;
  for (int t = 0; t < nk; ++t) {
    __syncthreads();                 // fill of `buf` visible; prev reads done
    if (t + 1 < nk) {
      fill_tile(buf ^ 1, (t + 1) << 6);
      if (t + 2 < nk) {              // prefetch tile t+2 (global_prefetch_b8)
        const int kp = (t + 2) << 6;
        __builtin_prefetch(TA ? &Aptr[(size_t)(kp + (tid & 31)) * lda + m0]
                              : &Aptr[(size_t)(m0 + (tid & 63)) * lda + kp], 0, 0);
        __builtin_prefetch(TB ? &Bptr[(size_t)(n0 + (tid & 63)) * ldb + kp]
                              : &Bptr[(size_t)(kp + (tid & 31)) * ldb + n0], 0, 0);
      }
    }
    const _Float16* Asb = &As[buf][0];
    const _Float16* Bsb = &Bs[buf][0];
#pragma unroll
    for (int ks = 0; ks < 2; ++ks) {
      v16h a[2], b[2];
#pragma unroll
      for (int q = 0; q < 2; ++q) {
        a[q] = ld_frag(Asb + (wr + 16 * q + mrow) * LDP + (ks << 5), half);
        b[q] = ld_frag(Bsb + (wc + 16 * q + mrow) * LDP + (ks << 5), half);
      }
#pragma unroll
      for (int tm = 0; tm < 2; ++tm)
#pragma unroll
        for (int tn = 0; tn < 2; ++tn)
          acc[tm][tn] = __builtin_amdgcn_wmma_f32_16x16x32_f16(
              false, a[tm], false, b[tn], (short)0, acc[tm][tn], false, false);
    }
    buf ^= 1;
  }

  // C/D layout: lane -> col = lane%16, row = v + 8*(lane/16) per 16x16 tile.
  const int ccol = lane & 15;
  const int rh   = lane >> 4;

  if (EPI == 0) {
#pragma unroll
    for (int tm = 0; tm < 2; ++tm)
#pragma unroll
      for (int tn = 0; tn < 2; ++tn) {
        const int gm0 = m0 + wr + 16 * tm;
        const int gn  = n0 + wc + 16 * tn + ccol;
#pragma unroll
        for (int v = 0; v < 8; ++v)
          Cptr[(size_t)(gm0 + v + 8 * rh) * Nc + gn] = acc[tm][tn][v];
      }
  } else if (EPI == 1) {
#pragma unroll
    for (int tm = 0; tm < 2; ++tm)
#pragma unroll
      for (int tn = 0; tn < 2; ++tn) {
        const int gm0 = m0 + wr + 16 * tm;
        const int gn  = n0 + wc + 16 * tn + ccol;
        const float bv = bias ? bias[gn] : 0.0f;
#pragma unroll
        for (int v = 0; v < 8; ++v) {
          const int gm = gm0 + v + 8 * rh;
          Cptr[(size_t)gm * Nc + gn] = acc[tm][tn][v] * rs[gm] + bv;
        }
      }
  } else {
    float lsum = 0.0f;
#pragma unroll
    for (int tm = 0; tm < 2; ++tm)
#pragma unroll
      for (int tn = 0; tn < 2; ++tn) {
        const int gm0 = m0 + wr + 16 * tm;
        const int gn  = n0 + wc + 16 * tn + ccol;
#pragma unroll
        for (int v = 0; v < 8; ++v) {
          const int gm = gm0 + v + 8 * rh;
          const float aval = (gm == gn) ? adjDiag[gm]
                                        : adjA[(size_t)gm * adjN + gn];
          const float d = aval - acc[tm][tn][v];
          lsum += d * d;
        }
      }
    red[tid] = lsum;
    __syncthreads();
    for (int s = 64; s > 0; s >>= 1) {
      if (tid < s) red[tid] += red[tid + s];
      __syncthreads();
    }
    if (tid == 0) atomicAdd(linkAcc, red[0]);
  }
}

// ---------------------------------------------------------------------------
// Elementwise / scatter / reduction kernels
// ---------------------------------------------------------------------------
__global__ void fill_kernel(float* p, float v, long n) {
  long i = (long)blockIdx.x * blockDim.x + threadIdx.x;
  if (i < n) p[i] = v;
}

__global__ void deg_scatter_kernel(const int* __restrict__ col, float* cnt, int E) {
  int e = blockIdx.x * blockDim.x + threadIdx.x;
  if (e < E) atomicAdd(&cnt[col[e]], 1.0f);
}

__global__ void rsqrt_inplace_kernel(float* p, int n) {
  int i = blockIdx.x * blockDim.x + threadIdx.x;
  if (i < n) p[i] = rsqrtf(fmaxf(p[i], 1e-12f));
}

__global__ void adj_scatter_kernel(const int* __restrict__ row, const int* __restrict__ col,
                                   float* A, int E, int N) {
  int e = blockIdx.x * blockDim.x + threadIdx.x;
  if (e < E) atomicAdd(&A[(size_t)row[e] * N + col[e]], 1.0f);
}

__global__ void diag_extract_set_kernel(float* A, float* diag, int N) {
  int i = blockIdx.x * blockDim.x + threadIdx.x;
  if (i < N) { diag[i] = A[(size_t)i * N + i]; A[(size_t)i * N + i] = 1.0f; }
}

__global__ void rowsum_disd_kernel(const float* __restrict__ A, float* disd, int N) {
  __shared__ float red[256];
  const int r = blockIdx.x, tid = threadIdx.x;
  float s = 0.0f;
  for (int j = tid; j < N; j += 256) s += A[(size_t)r * N + j];
  red[tid] = s; __syncthreads();
  for (int w = 128; w > 0; w >>= 1) { if (tid < w) red[tid] += red[tid + w]; __syncthreads(); }
  if (tid == 0) disd[r] = rsqrtf(fmaxf(red[0], 1e-12f));
}

__global__ void gcn_init_kernel(const float* __restrict__ XW, const float* __restrict__ dis,
                                const float* __restrict__ b, float* out, int N, int H) {
  long idx = (long)blockIdx.x * blockDim.x + threadIdx.x;
  if (idx >= (long)N * H) return;
  int i = (int)(idx / H), h = (int)(idx - (long)i * H);
  out[idx] = XW[idx] * dis[i] * dis[i] + b[h];
}

__global__ void gcn_scatter_kernel(const int* __restrict__ row, const int* __restrict__ col,
                                   const float* __restrict__ XW, const float* __restrict__ dis,
                                   float* out, int E, int H) {
  int idx = blockIdx.x * blockDim.x + threadIdx.x;
  const int hq = H >> 2;
  if (idx >= E * hq) return;
  int e = idx / hq, h4 = (idx - e * hq) << 2;
  int r = row[e], c = col[e];
  float w = dis[r] * dis[c];
  const float4 v = *(const float4*)(XW + (size_t)r * H + h4);
  float* o = out + (size_t)c * H + h4;
  atomicAdd(o + 0, v.x * w); atomicAdd(o + 1, v.y * w);
  atomicAdd(o + 2, v.z * w); atomicAdd(o + 3, v.w * w);
}

__global__ void tanh_inplace_kernel(float* p, long n) {
  long i = (long)blockIdx.x * blockDim.x + threadIdx.x;
  if (i < n) p[i] = tanhf(p[i]);
}

// rowwise softmax in place + fused entropy accumulation
__global__ void softmax_ent_kernel(float* S, float* entAcc, int K) {
  __shared__ float red[256];
  const int tid = threadIdx.x;
  float* rowp = S + (size_t)blockIdx.x * K;
  float m = -3.4e38f;
  for (int k = tid; k < K; k += 256) m = fmaxf(m, rowp[k]);
  red[tid] = m; __syncthreads();
  for (int w = 128; w > 0; w >>= 1) { if (tid < w) red[tid] = fmaxf(red[tid], red[tid + w]); __syncthreads(); }
  m = red[0]; __syncthreads();
  float s = 0.0f;
  for (int k = tid; k < K; k += 256) s += __expf(rowp[k] - m);
  red[tid] = s; __syncthreads();
  for (int w = 128; w > 0; w >>= 1) { if (tid < w) red[tid] += red[tid + w]; __syncthreads(); }
  const float inv = 1.0f / red[0]; __syncthreads();
  float ent = 0.0f;
  for (int k = tid; k < K; k += 256) {
    float p = __expf(rowp[k] - m) * inv;
    rowp[k] = p;
    ent -= p * logf(p + 1e-15f);
  }
  red[tid] = ent; __syncthreads();
  for (int w = 128; w > 0; w >>= 1) { if (tid < w) red[tid] += red[tid + w]; __syncthreads(); }
  if (tid == 0) atomicAdd(entAcc, red[0]);
}

// tmp[i,k] += (Adiag[i]-1) * ss[i,k]   (recover A@ss from A_sl@ss)
__global__ void tmp_fix_kernel(float* tmp, const float* __restrict__ ss,
                               const float* __restrict__ diag, int N, int K) {
  long idx = (long)blockIdx.x * blockDim.x + threadIdx.x;
  if (idx >= (long)N * K) return;
  int i = (int)(idx / K);
  tmp[idx] += (diag[i] - 1.0f) * ss[idx];
}

__global__ void finalize_kernel(const float* scal, float* out2) {
  out2[0] = 0.1f * sqrtf(scal[0]) / ((float)N_NODES * (float)N_NODES);
  out2[1] = 0.1f * (scal[1] / (float)N_NODES);
}

// ---------------------------------------------------------------------------
// Host-side helpers
// ---------------------------------------------------------------------------
template<int TA, int TB, int EPI>
static void launch_gemm(const float* A, const float* B, float* C, int M, int Nc, int Kd,
                        const float* rs, const float* bias,
                        const float* adjA, const float* adjDiag, float* acc, int adjN,
                        hipStream_t s) {
  dim3 grid(Nc / 64, M / 64);
  gemm_wmma_kernel<TA, TB, EPI><<<grid, 128, 0, s>>>(A, B, C, M, Nc, Kd, rs, bias,
                                                     adjA, adjDiag, acc, adjN);
}

static inline int nblk(long n, int t) { return (int)((n + t - 1) / t); }

// one sparse GCNConv: out = scatter(norm * (xin@W)) + b ; optional tanh
static void run_gcn(const float* xin, int Hin, const float* W, const float* b, int Hout,
                    const int* row, const int* col, int E, const float* diss,
                    float* XW, float* out, bool act, hipStream_t s) {
  launch_gemm<0, 0, 0>(xin, W, XW, N_NODES, Hout, Hin, nullptr, nullptr,
                       nullptr, nullptr, nullptr, 0, s);
  gcn_init_kernel<<<nblk((long)N_NODES * Hout, 256), 256, 0, s>>>(XW, diss, b, out, N_NODES, Hout);
  gcn_scatter_kernel<<<nblk((long)E * (Hout >> 2), 256), 256, 0, s>>>(row, col, XW, diss, out, E, Hout);
  if (act) tanh_inplace_kernel<<<nblk((long)N_NODES * Hout, 256), 256, 0, s>>>(out, (long)N_NODES * Hout);
}

extern "C" void kernel_launch(void* const* d_in, const int* in_sizes, int n_in,
                              void* d_out, int out_size, void* d_ws, size_t ws_size,
                              hipStream_t stream) {
  const float* nodes = (const float*)d_in[0];
  const int*   edges = (const int*)d_in[1];
  const float* W1 = (const float*)d_in[3];  const float* b1 = (const float*)d_in[4];
  const float* W2 = (const float*)d_in[5];  const float* b2 = (const float*)d_in[6];
  const float* Wp = (const float*)d_in[7];  const float* bp = (const float*)d_in[8];
  const float* W3 = (const float*)d_in[9];  const float* b3 = (const float*)d_in[10];
  const float* W4 = (const float*)d_in[11]; const float* b4 = (const float*)d_in[12];
  const float* W5 = (const float*)d_in[13]; const float* b5 = (const float*)d_in[14];

  const int E = in_sizes[1] / 2;
  const int* row = edges;
  const int* col = edges + E;

  // ---- workspace layout (floats), ~131 MB total ----
  float* ws   = (float*)d_ws;
  float* A    = ws;                                 // [N,N] dense adj (-> A_sl)
  float* S    = A    + (size_t)N_NODES * N_NODES;   // [N,K] s_pre -> softmax ss
  float* U    = S    + (size_t)N_NODES * KCL;       // [N,K] xwp -> tmp -> T2
  float* XP   = U    + (size_t)N_NODES * KCL;       // [K,H]
  float* ADJP = XP   + (size_t)KCL * HIDD;          // [K,K]
  float* X1   = ADJP + (size_t)KCL * KCL;           // [N,H]
  float* X2   = X1   + (size_t)N_NODES * HIDD;
  float* XOUT = X2   + (size_t)N_NODES * HIDD;
  float* X3   = XOUT + (size_t)N_NODES * HIDD;
  float* X4   = X3   + (size_t)N_NODES * HIDD;
  float* XW   = X4   + (size_t)N_NODES * HIDD;      // [N,H] GEMM staging
  float* DISS = XW   + (size_t)N_NODES * HIDD;      // [N]
  float* DISD = DISS + N_NODES;                     // [N]
  float* ADIAG= DISD + N_NODES;                     // [N]
  float* SCAL = ADIAG + N_NODES;                    // [2]: link, ent

  float* out_x    = (float*)d_out;                          // [N,FEAT]
  float* out_adj  = out_x + (size_t)N_NODES * FEATD;        // [N,N]
  float* out_scal = out_adj + (size_t)N_NODES * N_NODES;    // [2]

  // ---- degrees (sparse, with self loops) ----
  fill_kernel<<<nblk(N_NODES, 256), 256, 0, stream>>>(DISS, 1.0f, N_NODES);
  deg_scatter_kernel<<<nblk(E, 256), 256, 0, stream>>>(col, DISS, E);
  rsqrt_inplace_kernel<<<nblk(N_NODES, 256), 256, 0, stream>>>(DISS, N_NODES);

  // ---- dense adjacency: A, diag saved, diag <- 1 (A becomes A_sl), dis_d ----
  fill_kernel<<<nblk((long)N_NODES * N_NODES, 256), 256, 0, stream>>>(A, 0.0f, (long)N_NODES * N_NODES);
  adj_scatter_kernel<<<nblk(E, 256), 256, 0, stream>>>(row, col, A, E, N_NODES);
  diag_extract_set_kernel<<<nblk(N_NODES, 256), 256, 0, stream>>>(A, ADIAG, N_NODES);
  rowsum_disd_kernel<<<N_NODES, 256, 0, stream>>>(A, DISD, N_NODES);
  fill_kernel<<<1, 32, 0, stream>>>(SCAL, 0.0f, 2);

  // ---- sparse GCN encoder ----
  run_gcn(nodes, FEATD, W1, b1, HIDD, row, col, E, DISS, XW, X1, true, stream);
  run_gcn(X1,    HIDD,  W2, b2, HIDD, row, col, E, DISS, XW, X2, true, stream);

  // ---- DenseGCNConv assignment: s = dis_d . A_sl . (dis_d . (x2@Wp)) + bp ----
  launch_gemm<0, 0, 1>(X2, Wp, U, N_NODES, KCL, HIDD, DISD, nullptr,
                       nullptr, nullptr, nullptr, 0, stream);          // U = dis_d * (x2@Wp)
  launch_gemm<0, 0, 1>(A, U, S, N_NODES, KCL, N_NODES, DISD, bp,
                       nullptr, nullptr, nullptr, 0, stream);          // S = dis_d*(A_sl@U) + bp
  softmax_ent_kernel<<<N_NODES, 256, 0, stream>>>(S, SCAL + 1, KCL);   // S <- ss, ent acc

  // ---- diff pool ----
  launch_gemm<1, 0, 0>(S, X2, XP, KCL, HIDD, N_NODES, nullptr, nullptr,
                       nullptr, nullptr, nullptr, 0, stream);          // xp = ss^T @ x2
  launch_gemm<0, 0, 0>(A, S, U, N_NODES, KCL, N_NODES, nullptr, nullptr,
                       nullptr, nullptr, nullptr, 0, stream);          // U = A_sl @ ss
  tmp_fix_kernel<<<nblk((long)N_NODES * KCL, 256), 256, 0, stream>>>(U, S, ADIAG, N_NODES, KCL);
  launch_gemm<1, 0, 0>(S, U, ADJP, KCL, KCL, N_NODES, nullptr, nullptr,
                       nullptr, nullptr, nullptr, 0, stream);          // adjp = ss^T @ (A@ss)
  launch_gemm<0, 1, 2>(S, S, nullptr, N_NODES, N_NODES, KCL, nullptr, nullptr,
                       A, ADIAG, SCAL, N_NODES, stream);               // link = ||A - ss ss^T||^2

  // ---- un-pool ----
  launch_gemm<0, 0, 0>(S, ADJP, U, N_NODES, KCL, KCL, nullptr, nullptr,
                       nullptr, nullptr, nullptr, 0, stream);          // U = ss @ adjp
  launch_gemm<0, 1, 0>(U, S, out_adj, N_NODES, N_NODES, KCL, nullptr, nullptr,
                       nullptr, nullptr, nullptr, 0, stream);          // adj_out = U @ ss^T
  launch_gemm<0, 0, 0>(S, XP, XOUT, N_NODES, HIDD, KCL, nullptr, nullptr,
                       nullptr, nullptr, nullptr, 0, stream);          // x_out = ss @ xp

  // ---- sparse GCN decoder ----
  run_gcn(XOUT, HIDD, W3, b3, HIDD,  row, col, E, DISS, XW, X3,    true,  stream);
  run_gcn(X3,   HIDD, W4, b4, HIDD,  row, col, E, DISS, XW, X4,    true,  stream);
  run_gcn(X4,   HIDD, W5, b5, FEATD, row, col, E, DISS, XW, out_x, false, stream);

  // ---- scalars: 0.1*link, 0.1*ent ----
  finalize_kernel<<<1, 1, 0, stream>>>(SCAL, out_scal);
}
